// ProbAttention_10883447128296
// MI455X (gfx1250) — matrix operand order repaired and checked by reading, not compiled, not run on hardware
//
#include <hip/hip_runtime.h>

#define B_ 4
#define N_ 2048
#define C_ 512
#define H_ 8
#define D_ 64
#define U_ 40
#define UP_ 48
#define SK_ 40

typedef __attribute__((ext_vector_type(16))) __bf16 v16bf;
typedef __attribute__((ext_vector_type(8)))  float  v8f;
typedef __attribute__((ext_vector_type(4)))  int    v4i;

#if defined(__has_builtin)
#if __has_builtin(__builtin_amdgcn_global_load_async_to_lds_b128) && \
    __has_builtin(__builtin_amdgcn_s_wait_asynccnt)
#define HAVE_ASYNC_LDS 1
#endif
#endif
#ifndef HAVE_ASYNC_LDS
#define HAVE_ASYNC_LDS 0
#endif

#if HAVE_ASYNC_LDS
typedef __attribute__((address_space(1))) v4i* gv4i_p;  // global (prints as __device__)
typedef __attribute__((address_space(3))) v4i* sv4i_p;  // LDS (prints as __shared__)
#endif

__device__ __forceinline__ unsigned short f2bf(float f) {
  union { float f; unsigned u; } c; c.f = f;
  unsigned u = c.u;
  unsigned r = (u + 0x7FFFu + ((u >> 16) & 1u)) >> 16;  // RNE
  return (unsigned short)r;
}
__device__ __forceinline__ float bf2f(unsigned short h) {
  union { unsigned u; float f; } c; c.u = ((unsigned)h) << 16;
  return c.f;
}

// 16B per-lane copy global -> LDS (async on CDNA5, vectorized fallback otherwise)
__device__ __forceinline__ void cp16(void* lds_dst, const void* gsrc) {
#if HAVE_ASYNC_LDS
  __builtin_amdgcn_global_load_async_to_lds_b128((gv4i_p)gsrc, (sv4i_p)lds_dst, 0, 0);
#else
  *(uint4*)lds_dst = *(const uint4*)gsrc;
#endif
}
__device__ __forceinline__ void cp_wait() {
#if HAVE_ASYNC_LDS
  __builtin_amdgcn_s_wait_asynccnt(0);
#endif
}

// A-fragment: 16x32 bf16 tile, row-major LDS (leading dim lda, 16B-aligned rows).
// lanes 0-15: M=lane, elems 0..7 -> K=8*half..+7, elems 8..15 -> K=16+8*half..+7
__device__ __forceinline__ v16bf load_A_frag(const unsigned short* a, int lda, int lane) {
  int m = lane & 15, hv = lane >> 4;
  const unsigned short* p = a + m * lda + 8 * hv;
  union { uint4 q[2]; v16bf v; } u;
  u.q[0] = *(const uint4*)p;
  u.q[1] = *(const uint4*)(p + 16);
  return u.v;
}
// B-fragment from N-major LDS tile: bt[n][k], leading dim ldk.
// lane n in 0..15: elems 0..7 -> K=8*half..+7, elems 8..15 -> K=16+8*half..+7
__device__ __forceinline__ v16bf load_B_fragT(const unsigned short* bt, int ldk, int lane) {
  int n = lane & 15, hv = lane >> 4;
  const unsigned short* p = bt + n * ldk + hv * 8;
  union { uint4 q[2]; v16bf v; } u;
  u.q[0] = *(const uint4*)p;
  u.q[1] = *(const uint4*)(p + 16);
  return u.v;
}
__device__ __forceinline__ v8f wmma_bf16(v16bf a, v16bf b, v8f c) {
  return __builtin_amdgcn_wmma_f32_16x16x32_bf16(false, a, false, b, (short)0, c, false, false);
}

// ---------------- fp32 -> bf16 ----------------
__global__ __launch_bounds__(256) void cvt_bf16_kernel(const float* __restrict__ src,
                                                       unsigned short* __restrict__ dst, int n) {
  int i = blockIdx.x * 256 + threadIdx.x;
  if (i < n) dst[i] = f2bf(src[i]);
}
// fp32 [K][N] -> bf16 transposed [N][K]  (for weights; K=N=C_)
__global__ __launch_bounds__(256) void cvtT_bf16_kernel(const float* __restrict__ src,
                                                        unsigned short* __restrict__ dst) {
  int idx = blockIdx.x * 256 + threadIdx.x;
  if (idx >= C_ * C_) return;
  int k = idx >> 9, n = idx & (C_ - 1);
  dst[n * C_ + k] = f2bf(src[idx]);
}

// ---------------- tiled bf16 WMMA GEMM: Y = X(MxC_) @ W + bias ----------------
// W supplied TRANSPOSED: Wt[n][k].
// mode 0: bf16 head-split [B][H][N][D]; mode 1: fp32 [M][C_]; mode 2: bf16 [B][H][D][N]
__global__ __launch_bounds__(256) void gemm_bf16_kernel(const unsigned short* __restrict__ X,
                                                        const unsigned short* __restrict__ Wt,
                                                        const float* __restrict__ bias,
                                                        void* __restrict__ out, int M, int mode) {
  __shared__ alignas(16) unsigned short As[64][32];
  __shared__ alignas(16) unsigned short Bt[128][32];   // N-major
  const int K = C_, Nc = C_;
  int tile_m = blockIdx.x * 64;
  int tile_n = blockIdx.y * 128;
  int tid  = threadIdx.x;
  int wave = tid >> 5, lane = tid & 31;
  int wm = wave >> 2;      // 0..1 : 32 rows each
  int wn = wave & 3;       // 0..3 : 32 cols each

  v8f acc[2][2] = {};
  for (int k0 = 0; k0 < K; k0 += 32) {
    {  // stage A: 64 rows x 4 x 16B
      int r = tid >> 2, seg = tid & 3;
      cp16(&As[r][seg * 8], &X[(size_t)(tile_m + r) * K + k0 + seg * 8]);
    }
#pragma unroll
    for (int it = 0; it < 2; ++it) {  // stage B: 128 rows x 4 x 16B
      int i = tid + it * 256;
      int r = i >> 2, seg = i & 3;
      cp16(&Bt[r][seg * 8], &Wt[(size_t)(tile_n + r) * K + k0 + seg * 8]);
    }
    if (k0 + 32 < K) {
      __builtin_prefetch(&X[(size_t)(tile_m + (tid >> 2)) * K + k0 + 32], 0, 1);
      __builtin_prefetch(&Wt[(size_t)(tile_n + (tid >> 1)) * K + k0 + 32], 0, 1);
    }
    cp_wait();
    __syncthreads();
    v16bf af0 = load_A_frag(&As[wm * 32][0],       32, lane);
    v16bf af1 = load_A_frag(&As[wm * 32 + 16][0],  32, lane);
    v16bf bf0 = load_B_fragT(&Bt[wn * 32][0],      32, lane);
    v16bf bf1 = load_B_fragT(&Bt[wn * 32 + 16][0], 32, lane);
    acc[0][0] = wmma_bf16(af0, bf0, acc[0][0]);
    acc[0][1] = wmma_bf16(af0, bf1, acc[0][1]);
    acc[1][0] = wmma_bf16(af1, bf0, acc[1][0]);
    acc[1][1] = wmma_bf16(af1, bf1, acc[1][1]);
    __syncthreads();
  }
  int hv = lane >> 4, nn = lane & 15;
#pragma unroll
  for (int mi = 0; mi < 2; ++mi)
#pragma unroll
    for (int ni = 0; ni < 2; ++ni)
#pragma unroll
      for (int r = 0; r < 8; ++r) {
        int row = tile_m + wm * 32 + mi * 16 + r + 8 * hv;
        int col = tile_n + wn * 32 + ni * 16 + nn;
        float val = acc[mi][ni][r] + bias[col];
        if (mode == 0) {
          int b = row >> 11, n = row & (N_ - 1);
          int h = col >> 6,  d = col & (D_ - 1);
          ((unsigned short*)out)[(((b * H_ + h) * N_) + n) * D_ + d] = f2bf(val);
        } else if (mode == 1) {
          ((float*)out)[(size_t)row * Nc + col] = val;
        } else {
          int b = row >> 11, n = row & (N_ - 1);
          int h = col >> 6,  d = col & (D_ - 1);
          ((unsigned short*)out)[(((b * H_ + h) * D_) + d) * N_ + n] = f2bf(val);
        }
      }
}

// ---------------- sampled QK^T sparsity measure ----------------
__global__ __launch_bounds__(32) void qks_kernel(const unsigned short* __restrict__ q,
                                                 const unsigned short* __restrict__ k,
                                                 const int* __restrict__ isamp,
                                                 float* __restrict__ Mm) {
  int bh = blockIdx.x >> 11;
  int l  = blockIdx.x & (N_ - 1);
  int lane = threadIdx.x;
  const unsigned short* qrow = q + ((size_t)bh * N_ + l) * D_;
  float mx = -1e30f, sm = 0.f;
  for (int s = lane; s < SK_; s += 32) {
    int j = isamp[l * SK_ + s];
    const unsigned short* krow = k + ((size_t)bh * N_ + j) * D_;
    float d = 0.f;
#pragma unroll
    for (int t = 0; t < D_; ++t) d += bf2f(qrow[t]) * bf2f(krow[t]);
    mx = fmaxf(mx, d);
    sm += d;
  }
  for (int off = 16; off; off >>= 1) {
    mx = fmaxf(mx, __shfl_xor(mx, off));
    sm += __shfl_xor(sm, off);
  }
  if (lane == 0) Mm[blockIdx.x] = mx - sm * (1.0f / (float)N_);
}

// ---------------- top-u iterative argmax per (b,h) ----------------
__global__ __launch_bounds__(256) void topk_kernel(const float* __restrict__ Mm,
                                                   int* __restrict__ Mtop) {
  int bh = blockIdx.x, tid = threadIdx.x;
  __shared__ float vals[N_];
  __shared__ float rv[256];
  __shared__ int   ri[256];
  for (int i = tid; i < N_; i += 256) vals[i] = Mm[bh * N_ + i];
  __syncthreads();
  for (int it = 0; it < U_; ++it) {
    float bv = -1e30f; int bi = 0;
    for (int i = tid; i < N_; i += 256) {
      float x = vals[i];
      if (x > bv) { bv = x; bi = i; }
    }
    rv[tid] = bv; ri[tid] = bi;
    __syncthreads();
    for (int s = 128; s; s >>= 1) {
      if (tid < s && rv[tid + s] > rv[tid]) { rv[tid] = rv[tid + s]; ri[tid] = ri[tid + s]; }
      __syncthreads();
    }
    if (tid == 0) { Mtop[bh * U_ + it] = ri[0]; vals[ri[0]] = -1e30f; }
    __syncthreads();
  }
}

// ---------------- scores = Qr @ K^T * scale  (WMMA) ----------------
__global__ __launch_bounds__(128) void scores_kernel(const unsigned short* __restrict__ q,
                                                     const unsigned short* __restrict__ k,
                                                     const int* __restrict__ Mtop,
                                                     float* __restrict__ scores) {
  int bh = blockIdx.x;
  int n0 = blockIdx.y * 64;
  int tid = threadIdx.x, w = tid >> 5, lane = tid & 31;
  __shared__ alignas(16) unsigned short Qs[UP_][D_];  // gathered queries
  __shared__ alignas(16) unsigned short Bt[64][D_];   // N-major: Bt[j][d] = k[n0+j][d]
  for (int i = tid; i < U_ * 8; i += 128) {           // 40 rows x 8 x 16B
    int r = i >> 3, seg = i & 7;
    int src = Mtop[bh * U_ + r];
    cp16(&Qs[r][seg * 8], &q[((size_t)bh * N_ + src) * D_ + seg * 8]);
  }
  for (int i = tid; i < (UP_ - U_) * D_; i += 128) {  // zero pad rows
    int r = U_ + (i >> 6), c = i & (D_ - 1);
    Qs[r][c] = 0;
  }
  for (int i = tid; i < 64 * 8; i += 128) {           // 64 K-rows x 8 x 16B
    int r = i >> 3, seg = i & 7;
    cp16(&Bt[r][seg * 8], &k[((size_t)bh * N_ + n0 + r) * D_ + seg * 8]);
  }
  cp_wait();
  __syncthreads();
  v8f acc[3] = {};
#pragma unroll
  for (int ks = 0; ks < 2; ++ks) {
    v16bf bf = load_B_fragT(&Bt[w * 16][ks * 32], D_, lane);
#pragma unroll
    for (int mi = 0; mi < 3; ++mi) {
      v16bf af = load_A_frag(&Qs[mi * 16][ks * 32], D_, lane);
      acc[mi] = wmma_bf16(af, bf, acc[mi]);
    }
  }
  const float scale = 0.125f;  // D^-0.5
  int hv = lane >> 4, nn = lane & 15;
#pragma unroll
  for (int mi = 0; mi < 3; ++mi)
#pragma unroll
    for (int r = 0; r < 8; ++r) {
      int row = mi * 16 + r + 8 * hv;
      int col = n0 + w * 16 + nn;
      scores[((size_t)bh * UP_ + row) * N_ + col] = acc[mi][r] * scale;
    }
}

// ---------------- row softmax -> bf16 attn ----------------
__global__ __launch_bounds__(256) void softmax_kernel(const float* __restrict__ scores,
                                                      unsigned short* __restrict__ attn) {
  int row = blockIdx.x;
  int u = row % UP_;
  int tid = threadIdx.x;
  unsigned short* arow = attn + (size_t)row * N_;
  if (u >= U_) {
    for (int i = tid; i < N_; i += 256) arow[i] = 0;
    return;
  }
  const float* srow = scores + (size_t)row * N_;
  __shared__ float red[256];
  float mx = -1e30f;
  for (int i = tid; i < N_; i += 256) mx = fmaxf(mx, srow[i]);
  red[tid] = mx; __syncthreads();
  for (int s = 128; s; s >>= 1) { if (tid < s) red[tid] = fmaxf(red[tid], red[tid + s]); __syncthreads(); }
  mx = red[0]; __syncthreads();
  float sm = 0.f;
  for (int i = tid; i < N_; i += 256) sm += __expf(srow[i] - mx);
  red[tid] = sm; __syncthreads();
  for (int s = 128; s; s >>= 1) { if (tid < s) red[tid] += red[tid + s]; __syncthreads(); }
  sm = red[0]; __syncthreads();
  float inv = 1.0f / sm;
  for (int i = tid; i < N_; i += 256) arow[i] = f2bf(__expf(srow[i] - mx) * inv);
}

// ---------------- upd = attn @ V  (WMMA; V transposed [B][H][D][N]) ----------------
__global__ __launch_bounds__(128) void upd_kernel(const unsigned short* __restrict__ attn,
                                                  const unsigned short* __restrict__ vT,
                                                  float* __restrict__ upd) {
  int bh = blockIdx.x;
  int tid = threadIdx.x, w = tid >> 5, lane = tid & 31;
  __shared__ alignas(16) unsigned short As2[UP_][32];
  __shared__ alignas(16) unsigned short Bt[D_][32];   // N-major: Bt[d][k]
  v8f acc[3] = {};
  for (int k0 = 0; k0 < N_; k0 += 32) {
    for (int i = tid; i < UP_ * 4; i += 128) {        // 48 rows x 4 x 16B
      int r = i >> 2, seg = i & 3;
      cp16(&As2[r][seg * 8], &attn[((size_t)bh * UP_ + r) * N_ + k0 + seg * 8]);
    }
    for (int i = tid; i < D_ * 4; i += 128) {         // 64 rows x 4 x 16B
      int r = i >> 2, seg = i & 3;
      cp16(&Bt[r][seg * 8], &vT[((size_t)bh * D_ + r) * N_ + k0 + seg * 8]);
    }
    cp_wait();
    __syncthreads();
    v16bf bf = load_B_fragT(&Bt[w * 16][0], 32, lane);
#pragma unroll
    for (int mi = 0; mi < 3; ++mi) {
      v16bf af = load_A_frag(&As2[mi * 16][0], 32, lane);
      acc[mi] = wmma_bf16(af, bf, acc[mi]);
    }
    __syncthreads();
  }
  int hv = lane >> 4, nn = lane & 15;
#pragma unroll
  for (int mi = 0; mi < 3; ++mi)
#pragma unroll
    for (int r = 0; r < 8; ++r) {
      int row = mi * 16 + r + 8 * hv;
      int col = w * 16 + nn;
      upd[((size_t)bh * UP_ + row) * D_ + col] = acc[mi][r];
    }
}

// ---------------- vmean[b,h,d] over transposed V ----------------
__global__ __launch_bounds__(64) void vmean_kernel(const unsigned short* __restrict__ vT,
                                                   float* __restrict__ vmean) {
  int bh = blockIdx.x, d = threadIdx.x;
  const unsigned short* row = vT + ((size_t)bh * D_ + d) * N_;
  float s = 0.f;
  for (int n = 0; n < N_; ++n) s += bf2f(row[n]);
  vmean[bh * D_ + d] = s * (1.0f / (float)N_);
}

// ---------------- ctx fill with broadcast mean (bf16 [B*N][C]) ----------------
__global__ __launch_bounds__(256) void ctxfill_kernel(const float* __restrict__ vmean,
                                                      unsigned short* __restrict__ ctx) {
  int idx = blockIdx.x * 256 + threadIdx.x;
  if (idx >= B_ * N_ * C_) return;
  int b = idx >> 20;
  int c = idx & (C_ - 1);
  int h = c >> 6, d = c & (D_ - 1);
  ctx[idx] = f2bf(vmean[(b * H_ + h) * D_ + d]);
}

// ---------------- scatter dominant rows into ctx ----------------
__global__ __launch_bounds__(256) void scatter_kernel(const int* __restrict__ Mtop,
                                                      const float* __restrict__ upd,
                                                      unsigned short* __restrict__ ctx) {
  int bh = blockIdx.x;
  int b = bh >> 3, h = bh & (H_ - 1);
  for (int i = threadIdx.x; i < U_ * D_; i += 256) {
    int u = i >> 6, d = i & (D_ - 1);
    int n = Mtop[bh * U_ + u];
    ctx[((size_t)(b * N_ + n)) * C_ + h * D_ + d] =
        f2bf(upd[((size_t)bh * UP_ + u) * D_ + d]);
  }
}

extern "C" void kernel_launch(void* const* d_in, const int* in_sizes, int n_in,
                              void* d_out, int out_size, void* d_ws, size_t ws_size,
                              hipStream_t stream) {
  (void)in_sizes; (void)n_in; (void)out_size; (void)ws_size;
  const float* x  = (const float*)d_in[0];
  const float* Wq = (const float*)d_in[1];
  const float* bq = (const float*)d_in[2];
  const float* Wk = (const float*)d_in[3];
  const float* bk = (const float*)d_in[4];
  const float* Wv = (const float*)d_in[5];
  const float* bv = (const float*)d_in[6];
  const float* Wo = (const float*)d_in[7];
  const float* bo = (const float*)d_in[8];
  const int*   isamp = (const int*)d_in[9];

  char* p = (char*)d_ws;
  auto alloc = [&](size_t bytes) -> char* {
    char* r = p;
    p += (bytes + 255) & ~(size_t)255;
    return r;
  };
  const size_t MR = (size_t)B_ * N_;
  unsigned short* xh   = (unsigned short*)alloc(MR * C_ * 2);
  unsigned short* wqT  = (unsigned short*)alloc((size_t)C_ * C_ * 2);
  unsigned short* wkT  = (unsigned short*)alloc((size_t)C_ * C_ * 2);
  unsigned short* wvT  = (unsigned short*)alloc((size_t)C_ * C_ * 2);
  unsigned short* woT  = (unsigned short*)alloc((size_t)C_ * C_ * 2);
  unsigned short* qh   = (unsigned short*)alloc(MR * C_ * 2);
  unsigned short* kh   = (unsigned short*)alloc(MR * C_ * 2);
  unsigned short* vT   = (unsigned short*)alloc(MR * C_ * 2);
  float*          Mm   = (float*)alloc((size_t)B_ * H_ * N_ * 4);
  int*            Mtop = (int*)alloc((size_t)B_ * H_ * U_ * 4);
  float*          sc   = (float*)alloc((size_t)B_ * H_ * UP_ * N_ * 4);
  unsigned short* attn = (unsigned short*)alloc((size_t)B_ * H_ * UP_ * N_ * 2);
  float*          updb = (float*)alloc((size_t)B_ * H_ * UP_ * D_ * 4);
  float*          vmn  = (float*)alloc((size_t)B_ * H_ * D_ * 4);
  unsigned short* ctx  = (unsigned short*)alloc(MR * C_ * 2);

  // 1) conversions (x plain, weights transposed)
  cvt_bf16_kernel<<<(MR * C_ + 255) / 256, 256, 0, stream>>>(x, xh, (int)(MR * C_));
  int wgrid = (C_ * C_ + 255) / 256;
  cvtT_bf16_kernel<<<wgrid, 256, 0, stream>>>(Wq, wqT);
  cvtT_bf16_kernel<<<wgrid, 256, 0, stream>>>(Wk, wkT);
  cvtT_bf16_kernel<<<wgrid, 256, 0, stream>>>(Wv, wvT);
  cvtT_bf16_kernel<<<wgrid, 256, 0, stream>>>(Wo, woT);

  // 2) projections (WMMA GEMM); V written transposed
  dim3 gg((unsigned)(MR / 64), C_ / 128);
  gemm_bf16_kernel<<<gg, 256, 0, stream>>>(xh, wqT, bq, qh, (int)MR, 0);
  gemm_bf16_kernel<<<gg, 256, 0, stream>>>(xh, wkT, bk, kh, (int)MR, 0);
  gemm_bf16_kernel<<<gg, 256, 0, stream>>>(xh, wvT, bv, vT, (int)MR, 2);

  // 3) sampled sparsity measure + top-u
  qks_kernel<<<B_ * H_ * N_, 32, 0, stream>>>(qh, kh, isamp, Mm);
  topk_kernel<<<B_ * H_, 256, 0, stream>>>(Mm, Mtop);

  // 4) reduced-Q attention (WMMA)
  scores_kernel<<<dim3(B_ * H_, N_ / 64), 128, 0, stream>>>(qh, kh, Mtop, sc);
  softmax_kernel<<<B_ * H_ * UP_, 256, 0, stream>>>(sc, attn);
  upd_kernel<<<B_ * H_, 128, 0, stream>>>(attn, vT, updb);

  // 5) context = mean(V) with dominant rows replaced
  vmean_kernel<<<B_ * H_, 64, 0, stream>>>(vT, vmn);
  ctxfill_kernel<<<(B_ * N_ * C_ + 255) / 256, 256, 0, stream>>>(vmn, ctx);
  scatter_kernel<<<B_ * H_, 256, 0, stream>>>(Mtop, updb, ctx);

  // 6) output projection (WMMA GEMM -> fp32 d_out)
  gemm_bf16_kernel<<<gg, 256, 0, stream>>>(ctx, woT, bo, (float*)d_out, (int)MR, 1);
}
